// Model_39676907881369
// MI455X (gfx1250) — matrix operand order repaired and checked by reading, not compiled, not run on hardware
//
#include <hip/hip_runtime.h>
#include <cstdint>

typedef _Float16 v16h __attribute__((ext_vector_type(16)));
typedef _Float16 h8   __attribute__((ext_vector_type(8)));
typedef float    v8f  __attribute__((ext_vector_type(8)));
typedef unsigned int u32x4 __attribute__((ext_vector_type(4)));
typedef int          i32x4 __attribute__((ext_vector_type(4)));
typedef int          i32x8 __attribute__((ext_vector_type(8)));

#define WMMA_F16(a, b, c) \
  __builtin_amdgcn_wmma_f32_16x16x32_f16(false, (a), false, (b), (short)0, (c), false, false)

// Deterministic per-element dropout hash (keep prob ~0.8).
__device__ __forceinline__ bool dropout_keep(unsigned x) {
  x = x * 747796405u + 2891336453u;
  unsigned w = ((x >> ((x >> 28) + 4u)) ^ x) * 277803737u;
  w = (w >> 22) ^ w;
  return (w & 0xFFFFFFu) < 13421773u;  // 0.8 * 2^24
}

// ---------------------------------------------------------------------------
// TDM: async DMA of a 32-row x 256-half fp16 tile (row stride 256 halves in
// memory) into LDS at lds_off, with 16B padding after each 512B row so the
// LDS pitch is 528B (= SKP 264 halves), matching the bank-conflict-free
// layout the WMMA B-operand reads expect.  Issued once per wave (EXEC
// ignored); tracked by TENSORcnt.
// ---------------------------------------------------------------------------
__device__ __forceinline__ void tdm_load_k_tile(unsigned lds_off,
                                                const _Float16* gptr) {
  unsigned long long ga = (unsigned long long)(uintptr_t)gptr;
  u32x4 g0;
  g0[0] = 1u;                                   // count=1 (valid user D#)
  g0[1] = lds_off;                              // lds_addr
  g0[2] = (unsigned)(ga & 0xFFFFFFFFull);       // global_addr[31:0]
  g0[3] = (unsigned)((ga >> 32) & 0x01FFFFFFull) | (2u << 30);  // [56:32], type=2
  i32x8 g1;
  // data_size=1 (2B) | pad_enable | pad_interval=6 (128 DW = 512B) |
  // pad_amount=3 (4 DW = 16B)
  g1[0] = (int)((1u << 16) | (1u << 20) | (6u << 22) | (3u << 25));
  g1[1] = (int)(256u << 16);    // tensor_dim0 = 256 elements (bits 63:48)
  g1[2] = (int)(16384u << 16);  // tensor_dim1 lo16 (bits 95:80)
  g1[3] = (int)(256u << 16);    // tile_dim0 = 256 (bits 127:112)
  g1[4] = 32;                   // tile_dim1 = 32 (bits 143:128), tile_dim2 = 0
  g1[5] = 256;                  // tensor_dim0_stride = 256 (bits 191:160)
  g1[6] = 0;
  g1[7] = 0;
  i32x4 gz = {0, 0, 0, 0};      // 2D tensor: groups 2/3 unused
#if defined(__clang_major__) && (__clang_major__ >= 23)
  i32x8 gz8 = {0, 0, 0, 0, 0, 0, 0, 0};
  __builtin_amdgcn_tensor_load_to_lds(g0, g1, gz, gz, gz8, 0);
#else
  __builtin_amdgcn_tensor_load_to_lds(g0, g1, gz, gz, 0);
#endif
}

// ---------------------------------------------------------------------------
// Projection: out[M,N] (fp16) = X[M,K] (fp32) @ W[K,N] (fp32) + bias[N]
// ---------------------------------------------------------------------------
template <int K, int N>
__global__ __launch_bounds__(256) void proj_wmma(const float* __restrict__ X,
                                                 const float* __restrict__ W,
                                                 const float* __restrict__ bias,
                                                 _Float16* __restrict__ out) {
  constexpr int KP = K + 8;
  __shared__ _Float16 sWT[16 * KP];
  const int tid = threadIdx.x;
  const int n0 = blockIdx.y * 16;
  for (int idx = tid; idx < 16 * K; idx += 256) {
    int k = idx >> 4, n = idx & 15;
    sWT[n * KP + k] = (_Float16)W[k * N + n0 + n];
  }
  __syncthreads();

  const int wave = tid >> 5, lane = tid & 31;
  const int row0 = (blockIdx.x * 8 + wave) * 16;
  const int m = lane & 15;
  const int koff = (lane < 16) ? 0 : 16;  // A-layout K sub-offset
  const int kb2  = (lane < 16) ? 0 : 16;  // B-layout K sub-offset

  v8f acc;
#pragma unroll
  for (int i = 0; i < 8; ++i) acc[i] = 0.f;

#pragma unroll
  for (int c = 0; c < K / 32; ++c) {
    const float* arow = X + (size_t)(row0 + m) * K + c * 32 + koff;
    v16h a;
#pragma unroll
    for (int e = 0; e < 8; ++e) {
      a[e]     = (_Float16)arow[e];
      a[e + 8] = (_Float16)arow[e + 8];
    }
    const _Float16* wp = &sWT[m * KP + c * 32 + kb2];
    v16h bm;
    *(h8*)&bm       = *(const h8*)wp;
    *((h8*)&bm + 1) = *(const h8*)(wp + 8);
    acc = WMMA_F16(a, bm, acc);
  }

  const float bv = bias[n0 + m];
#pragma unroll
  for (int r = 0; r < 8; ++r) {
    int lr = r + ((lane < 16) ? 0 : 8);  // C-layout row mapping
    out[(size_t)(row0 + lr) * N + n0 + m] = (_Float16)(acc[r] + bv);
  }
}

// ---------------------------------------------------------------------------
// Flash attention, TDM double-buffered K staging + register-prefetched V.
// Block = 128 threads (4 waves); grid = (Lq/64, B).
// ---------------------------------------------------------------------------
__global__ __launch_bounds__(128) void flash_attn(const _Float16* __restrict__ qp,
                                                  const _Float16* __restrict__ kp,
                                                  const _Float16* __restrict__ vp,
                                                  float* __restrict__ out) {
  constexpr int L = 4096, D = 256, DV = 64;
  constexpr int NIT = L / 32;
  constexpr float SCALE = 0.0625f;  // 1/sqrt(256)
  constexpr float LOG2E = 1.4426950408889634f;
  constexpr int SKP = 264;  // K-tile pitch (halves): 528B (512B row + 16B TDM pad)
  constexpr int SVP = 40;   // V^T row pitch: 80B
  constexpr int SPP = 40;   // P staging row pitch: 80B
  __shared__ _Float16 sK[2 * 32 * SKP];
  __shared__ _Float16 sVT[2 * 64 * SVP];
  __shared__ _Float16 sP[4 * 16 * SPP];

  const int tid = threadIdx.x;
  const int wave = tid >> 5, lane = tid & 31;
  const int b = blockIdx.y;
  const int qrow0 = blockIdx.x * 64 + wave * 16;
  const int m = lane & 15;
  const int koff  = (lane < 16) ? 0 : 16;
  const int kb2   = (lane < 16) ? 0 : 16;
  const int rhalf = (lane < 16) ? 0 : 8;

  const unsigned skoff[2] = {(unsigned)(uintptr_t)&sK[0],
                             (unsigned)(uintptr_t)&sK[32 * SKP]};
  const int vrow = tid >> 2;        // 0..31 key row handled by this thread
  const int vd0  = (tid & 3) * 16;  // 0,16,32,48 column group

  // Q tile into registers, pre-packed as WMMA A operands (8 chunks of K=32)
  v16h aq[8];
  {
    const _Float16* qrow = qp + (size_t)(b * L + qrow0 + m) * D;
#pragma unroll
    for (int c = 0; c < 8; ++c) {
      *(h8*)&aq[c]       = *(const h8*)(qrow + c * 32 + koff);
      *((h8*)&aq[c] + 1) = *(const h8*)(qrow + c * 32 + koff + 8);
    }
  }

  v8f Oacc[4];
#pragma unroll
  for (int t = 0; t < 4; ++t)
#pragma unroll
    for (int i = 0; i < 8; ++i) Oacc[t][i] = 0.f;
  float mrow[8], lrow[8];
#pragma unroll
  for (int r = 0; r < 8; ++r) { mrow[r] = -3.0e38f; lrow[r] = 0.f; }

  // ---- prologue: stage tile 0 --------------------------------------------
  if (wave == 0) tdm_load_k_tile(skoff[0], kp + (size_t)(b * L) * D);
  {
    const h8* src = (const h8*)(vp + (size_t)(b * L + vrow) * DV + vd0);
    h8 v0 = src[0], v1 = src[1];
#pragma unroll
    for (int e = 0; e < 8; ++e) {
      sVT[(vd0 + e) * SVP + vrow]     = v0[e];
      sVT[(vd0 + 8 + e) * SVP + vrow] = v1[e];
    }
  }
  if (wave == 0) __builtin_amdgcn_s_wait_tensorcnt(0);
  __syncthreads();

  for (int it = 0; it < NIT; ++it) {
    const int kb = it * 32;
    const int cur = it & 1, nxt = cur ^ 1;
    const _Float16* sKc  = sK + cur * (32 * SKP);
    const _Float16* sVTc = sVT + cur * (64 * SVP);

    // ---- kick off next tile's staging (async under compute) -------------
    h8 nv0, nv1;
    const bool more = (it + 1) < NIT;
    if (more) {
      if (wave == 0)
        tdm_load_k_tile(skoff[nxt], kp + (size_t)(b * L + kb + 32) * D);
      const h8* src = (const h8*)(vp + (size_t)(b * L + kb + 32 + vrow) * DV + vd0);
      nv0 = src[0];
      nv1 = src[1];
    }

    // ---- S = Q K^T for two 16-key sub-tiles ------------------------------
    v8f S0, S1;
#pragma unroll
    for (int i = 0; i < 8; ++i) { S0[i] = 0.f; S1[i] = 0.f; }
#pragma unroll
    for (int c = 0; c < 8; ++c) {
      const _Float16* p0 = sKc + m * SKP + c * 32 + kb2;
      const _Float16* p1 = sKc + (16 + m) * SKP + c * 32 + kb2;
      v16h b0, b1;
      *(h8*)&b0       = *(const h8*)p0;
      *((h8*)&b0 + 1) = *(const h8*)(p0 + 8);
      *(h8*)&b1       = *(const h8*)p1;
      *((h8*)&b1 + 1) = *(const h8*)(p1 + 8);
      S0 = WMMA_F16(aq[c], b0, S0);
      S1 = WMMA_F16(aq[c], b1, S1);
    }

    // ---- online softmax + dropout, stage P to LDS ------------------------
#pragma unroll
    for (int r = 0; r < 8; ++r) {
      float s0 = S0[r] * SCALE, s1 = S1[r] * SCALE;
      float mx = fmaxf(s0, s1);
#pragma unroll
      for (int off = 1; off < 16; off <<= 1) mx = fmaxf(mx, __shfl_xor(mx, off, 32));
      float mnew = fmaxf(mrow[r], mx);
      float resc = exp2f((mrow[r] - mnew) * LOG2E);
      Oacc[0][r] *= resc; Oacc[1][r] *= resc;
      Oacc[2][r] *= resc; Oacc[3][r] *= resc;
      float e0 = exp2f((s0 - mnew) * LOG2E);
      float e1 = exp2f((s1 - mnew) * LOG2E);
      float rs = e0 + e1;
#pragma unroll
      for (int off = 1; off < 16; off <<= 1) rs += __shfl_xor(rs, off, 32);
      lrow[r] = lrow[r] * resc + rs;  // denominator uses UNMASKED exp (ref order)
      mrow[r] = mnew;
      int lr = r + rhalf;
      unsigned base = ((unsigned)(b * L + qrow0 + lr) << 12) + (unsigned)(kb + m);
      if (!dropout_keep(base)) e0 = 0.f;
      if (!dropout_keep(base + 16u)) e1 = 0.f;
      sP[wave * (16 * SPP) + lr * SPP + m]      = (_Float16)e0;
      sP[wave * (16 * SPP) + lr * SPP + 16 + m] = (_Float16)e1;
    }

    // ---- P (A-layout via LDS) times V ------------------------------------
    {
      const _Float16* pp = sP + wave * (16 * SPP) + m * SPP + koff;
      v16h pa;
      *(h8*)&pa       = *(const h8*)pp;
      *((h8*)&pa + 1) = *(const h8*)(pp + 8);
#pragma unroll
      for (int t = 0; t < 4; ++t) {
        const _Float16* vpp = sVTc + (t * 16 + m) * SVP + kb2;
        v16h bvv;
        *(h8*)&bvv       = *(const h8*)vpp;
        *((h8*)&bvv + 1) = *(const h8*)(vpp + 8);
        Oacc[t] = WMMA_F16(pa, bvv, Oacc[t]);
      }
    }

    // ---- commit next tile's V^T, wait K DMA, single barrier --------------
    if (more) {
      _Float16* sVTn = sVT + nxt * (64 * SVP);
#pragma unroll
      for (int e = 0; e < 8; ++e) {
        sVTn[(vd0 + e) * SVP + vrow]     = nv0[e];
        sVTn[(vd0 + 8 + e) * SVP + vrow] = nv1[e];
      }
      if (wave == 0) __builtin_amdgcn_s_wait_tensorcnt(0);
    }
    __syncthreads();
  }

  // ---- epilogue: normalize by l and dropout keep-prob ----------------------
#pragma unroll
  for (int r = 0; r < 8; ++r) {
    float inv = 1.f / (lrow[r] * 0.8f);
    int lr = r + rhalf;
    float* orow = out + (size_t)(b * L + qrow0 + lr) * DV;
#pragma unroll
    for (int t = 0; t < 4; ++t) orow[t * 16 + m] = Oacc[t][r] * inv;
  }
}

// ---------------------------------------------------------------------------
extern "C" void kernel_launch(void* const* d_in, const int* in_sizes, int n_in,
                              void* d_out, int out_size, void* d_ws, size_t ws_size,
                              hipStream_t stream) {
  const float* q  = (const float*)d_in[0];
  const float* k  = (const float*)d_in[1];
  const float* v  = (const float*)d_in[2];
  const float* Wq = (const float*)d_in[3];
  const float* bq = (const float*)d_in[4];
  const float* Wk = (const float*)d_in[5];
  const float* bk = (const float*)d_in[6];
  const float* Wv = (const float*)d_in[7];
  const float* bv = (const float*)d_in[8];
  float* out = (float*)d_out;

  char* ws = (char*)d_ws;
  _Float16* qp = (_Float16*)ws;                              // 8 MB
  _Float16* kp = (_Float16*)(ws + (size_t)16384 * 256 * 2);  // 8 MB
  _Float16* vp = (_Float16*)(ws + (size_t)16384 * 256 * 4);  // 2 MB

  proj_wmma<256, 256><<<dim3(128, 16), 256, 0, stream>>>(q, Wq, bq, qp);
  proj_wmma<32, 256><<<dim3(128, 16), 256, 0, stream>>>(k, Wk, bk, kp);
  proj_wmma<256, 64><<<dim3(128, 4), 256, 0, stream>>>(v, Wv, bv, vp);

  flash_attn<<<dim3(64, 4), 128, 0, stream>>>(qp, kp, vp, out);
}
